// NonSymmetricAttentionKernel_89455578841178
// MI455X (gfx1250) — compile-verified, matrix operand order checked by reading, not run on hardware
//
#include <hip/hip_runtime.h>

#define B_ 2
#define S_ 2048
#define E_ 768
#define H_ 12
#define D_ 64

typedef __bf16 bf16_t;
typedef bf16_t bf16x8  __attribute__((ext_vector_type(8)));
typedef bf16_t bf16x16 __attribute__((ext_vector_type(16)));
typedef float  f32x8   __attribute__((ext_vector_type(8)));

#define WMMA(A, Bm, C) \
  __builtin_amdgcn_wmma_f32_16x16x32_bf16(false, (A), false, (Bm), (short)0, (C), false, false)

// Wait until ASYNCcnt <= n (n must be an immediate).
#define S_WAIT_ASYNCCNT(n) asm volatile("s_wait_asynccnt %0" :: "i"(n) : "memory")

static __device__ __forceinline__ unsigned short f2bf_bits(float f) {
  union { float f; unsigned u; } x; x.f = f;
  unsigned r = (x.u + 0x7FFFu + ((x.u >> 16) & 1u)) >> 16;  // RNE
  return (unsigned short)r;
}
static __device__ __forceinline__ bf16_t f2bf(float f) {
  union { unsigned short s; bf16_t b; } y; y.s = f2bf_bits(f); return y.b;
}

// Async DMA of 16 bytes per lane: global -> LDS, tracked with ASYNCcnt.
// LDS operand is the low 32 bits of the generic shared-memory address
static __device__ __forceinline__ void async_copy_b128(void* lds, const void* gmem) {
  asm volatile("global_load_async_to_lds_b128 %0, %1, off"
               :: "v"((unsigned)(uintptr_t)lds),
                  "v"((unsigned long long)(uintptr_t)gmem)
               : "memory");
}

// Load one 16x32 bf16 WMMA fragment (A or B operand) from a row-major tile.
// Per ISA 16-bit A layout: lane L holds row (L&15); K offsets {koff..koff+7} and
// {koff+16..koff+23} with koff = (L<16 ? 0 : 8)  ->  two 16-byte loads per lane.
static __device__ __forceinline__ bf16x16 frag_ld(const bf16_t* p, int stride, int kbase) {
  const int lane = threadIdx.x & 31;
  const int idx  = lane & 15;
  const int koff = (lane >> 4) << 3;
  const bf16_t* q = p + idx * stride + kbase + koff;
  union { bf16x16 v; bf16x8 h[2]; } u;
  u.h[0] = *(const bf16x8*)(q);
  u.h[1] = *(const bf16x8*)(q + 16);
  return u.v;
}

// ---------------------------------------------------------------------------
// Kernel 1: fused QKV projection.  X[4096x768] * W[768x768] + b, bf16 output.
// z=0 -> Q [B,H,S,D], z=1 -> K [B,H,S,D], z=2 -> V transposed [B,H,D,S].
// Block = 256 threads (8 waves); tile 128(M) x 64(N); each wave does 32x32.
// ---------------------------------------------------------------------------
__global__ __launch_bounds__(256) void qkv_gemm_kernel(
    const float* __restrict__ X,
    const float* __restrict__ Wq, const float* __restrict__ bq,
    const float* __restrict__ Wk, const float* __restrict__ bk,
    const float* __restrict__ Wv, const float* __restrict__ bv,
    bf16_t* __restrict__ q_out, bf16_t* __restrict__ k_out,
    bf16_t* __restrict__ vt_out) {
  __shared__ __align__(16) bf16_t Xs[128][40];  // 128x32 tile, padded
  __shared__ __align__(16) bf16_t Ws[64][40];   // 32x64 tile, stored [n][k]
  __shared__ __align__(16) bf16_t Os[128][72];  // output staging (Q/K path)

  const int tid  = threadIdx.x;
  const int lane = tid & 31;
  const int wave = tid >> 5;
  const int z  = blockIdx.z;
  const int m0 = blockIdx.x * 128;
  const int n0 = blockIdx.y * 64;
  const int wave_m = (wave & 3) * 32;
  const int wave_n = (wave >> 2) * 32;
  const int halfoff = (lane >> 4) << 3;
  const int nlan = lane & 15;

  const float* W    = (z == 0) ? Wq : (z == 1) ? Wk : Wv;
  const float* bias = (z == 0) ? bq : (z == 1) ? bk : bv;

  f32x8 acc[2][2] = {};

  for (int kb = 0; kb < E_; kb += 32) {
    // stage X tile (f32 -> bf16): 2048 float2 / 256 threads
    #pragma unroll
    for (int j = 0; j < 8; ++j) {
      int p = tid + 256 * j;
      int r = p >> 4, kp = p & 15;
      const float2 xv = *(const float2*)(X + (size_t)(m0 + r) * E_ + kb + kp * 2);
      unsigned pk = ((unsigned)f2bf_bits(xv.y) << 16) | (unsigned)f2bf_bits(xv.x);
      *(unsigned*)&Xs[r][kp * 2] = pk;
    }
    // stage W tile transposed [n][k]: float2 along n, 1024 pairs / 256 threads
    #pragma unroll
    for (int i = 0; i < 4; ++i) {
      int p = tid + 256 * i;
      int k = p >> 5, n2 = (p & 31) * 2;
      const float2 wv = *(const float2*)(W + (size_t)(kb + k) * E_ + n0 + n2);
      Ws[n2][k]     = f2bf(wv.x);
      Ws[n2 + 1][k] = f2bf(wv.y);
    }
    if (kb + 32 < E_)
      __builtin_prefetch(W + (size_t)(kb + 32 + (tid >> 5)) * E_ + n0 + (tid & 31) * 2, 0, 1);
    __syncthreads();

    bf16x16 a0 = frag_ld(&Xs[wave_m][0],      40, 0);
    bf16x16 a1 = frag_ld(&Xs[wave_m + 16][0], 40, 0);
    bf16x16 b0 = frag_ld(&Ws[wave_n][0],      40, 0);
    bf16x16 b1 = frag_ld(&Ws[wave_n + 16][0], 40, 0);
    acc[0][0] = WMMA(a0, b0, acc[0][0]);
    acc[0][1] = WMMA(a0, b1, acc[0][1]);
    acc[1][0] = WMMA(a1, b0, acc[1][0]);
    acc[1][1] = WMMA(a1, b1, acc[1][1]);
    __syncthreads();
  }

  const int hh = n0 >> 6;                 // whole block lies inside one head
  const size_t bh_base = (size_t)hh;      // head index within [b]

  if (z == 2) {
    // V path: C layout gives each lane 8 consecutive s positions at fixed d
    // -> packed bf16x8 stores straight from the accumulators (4 per lane).
    #pragma unroll
    for (int mi = 0; mi < 2; ++mi) {
      #pragma unroll
      for (int ni = 0; ni < 2; ++ni) {
        int dd = wave_n + ni * 16 + nlan;           // d within head
        float bval = bias[n0 + dd];
        int mg0 = m0 + wave_m + mi * 16 + halfoff;  // first of 8 consecutive rows
        int bidx = mg0 >> 11, ss0 = mg0 & (S_ - 1);
        bf16x8 pk;
        #pragma unroll
        for (int r = 0; r < 8; ++r) pk[r] = f2bf(acc[mi][ni][r] + bval);
        *(bf16x8*)(vt_out + (((size_t)bidx * H_ + bh_base) * D_ + dd) * S_ + ss0) = pk;
      }
    }
  } else {
    // Q/K path: stage the 128x64 tile in LDS, then coalesced b128 stores.
    bf16_t* dst = (z == 1) ? k_out : q_out;
    #pragma unroll
    for (int mi = 0; mi < 2; ++mi) {
      #pragma unroll
      for (int ni = 0; ni < 2; ++ni) {
        int cl = wave_n + ni * 16 + nlan;
        float bval = bias[n0 + cl];
        #pragma unroll
        for (int r = 0; r < 8; ++r)
          Os[wave_m + mi * 16 + r + halfoff][cl] = f2bf(acc[mi][ni][r] + bval);
      }
    }
    __syncthreads();
    #pragma unroll
    for (int i = 0; i < 4; ++i) {
      int c = tid + 256 * i;                 // 1024 x 16B chunks
      int row = c >> 3, ch = c & 7;
      int mg = m0 + row;
      int bidx = mg >> 11, ss = mg & (S_ - 1);
      *(bf16x8*)(dst + (((size_t)bidx * H_ + bh_base) * S_ + ss) * D_ + ch * 8) =
          *(const bf16x8*)&Os[row][ch * 8];
    }
  }
}

// Stage one 64x64 bf16 tile (8 KB) into LDS with 4 async B128 ops per thread-
// group pass: 128 threads x 4 chunks x 16 B.  Rows padded to 72 elems.
static __device__ __forceinline__ void stage_tile_async(
    bf16_t (*dst)[72], const bf16_t* gbase, int rowstride) {
  const int tid = threadIdx.x;
  #pragma unroll
  for (int i = 0; i < 4; ++i) {
    int c = tid + 128 * i;          // chunk id in [0,512)
    int row = c >> 3, ch = c & 7;   // 8 x 16B chunks per 64-elem row
    async_copy_b128(&dst[row][ch * 8], gbase + (size_t)row * rowstride + ch * 8);
  }
}

// ---------------------------------------------------------------------------
// Kernel 2: flash-style ALiBi attention with double-buffered async K/V tiles.
// grid = (S/64, H, B); block = 128 (4 waves); each wave owns 16 query rows.
// Head groups: h<4 causal (mask n>m), 4<=h<8 anti-causal (mask n<m), h>=8 none.
// ---------------------------------------------------------------------------
__global__ __launch_bounds__(128) void alibi_attn_kernel(
    const bf16_t* __restrict__ Q, const bf16_t* __restrict__ K,
    const bf16_t* __restrict__ Vt, float* __restrict__ out) {
  __shared__ __align__(16) bf16_t Ks[2][64][72];  // K tile:  [key_local][d]
  __shared__ __align__(16) bf16_t Vs[2][64][72];  // Vt tile: [d][key_local]
  __shared__ __align__(16) bf16_t Ps[4][16][72];  // per-wave P (C->A relayout)

  const int lane = threadIdx.x & 31;
  const int wave = threadIdx.x >> 5;
  const int qt = blockIdx.x;
  const int h  = blockIdx.y;
  const int b  = blockIdx.z;
  const int grp = h >> 2;
  const int q0_blk = qt * 64;
  const int q0 = q0_blk + wave * 16;
  const int halfoff = (lane >> 4) << 3;
  const int nlan = lane & 15;

  const float slope = -exp2f(-2.0f * (float)((h & 3) + 1));  // -0.25^((h%4)+1)
  const size_t bh = (size_t)b * H_ + h;
  const bf16_t* Qbh = Q  + bh * S_ * D_;
  const bf16_t* Kbh = K  + bh * S_ * D_;
  const bf16_t* Vbh = Vt + bh * D_ * S_;

  // Q fragments live in registers for the whole kernel
  bf16x16 qa0 = frag_ld(Qbh + (size_t)q0 * D_, D_, 0);
  bf16x16 qa1 = frag_ld(Qbh + (size_t)q0 * D_, D_, 32);

  float sqm[8]; int mg[8];
  #pragma unroll
  for (int r = 0; r < 8; ++r) {
    mg[r] = q0 + r + halfoff;
    sqm[r] = sqrtf((float)mg[r]);
  }

  f32x8 o[4] = {};
  float m_r[8], l_r[8];
  #pragma unroll
  for (int r = 0; r < 8; ++r) { m_r[r] = -1e30f; l_r[r] = 0.0f; }

  const int k_lo = (grp == 1) ? q0_blk : 0;
  const int k_hi = (grp == 0) ? (q0_blk + 64) : S_;
  const int nt   = (k_hi - k_lo) >> 6;  // >= 1 always

  // prologue: stage tile 0 (8 async ops per wave)
  stage_tile_async(Ks[0], Kbh + (size_t)k_lo * D_, D_);
  stage_tile_async(Vs[0], Vbh + k_lo, S_);

  for (int t = 0; t < nt; ++t) {
    const int kt  = k_lo + t * 64;
    const int cur = t & 1;
    if (t + 1 < nt) {
      __syncthreads();  // all waves done reading buf[cur^1] (iteration t-1)
      stage_tile_async(Ks[cur ^ 1], Kbh + (size_t)(kt + 64) * D_, D_);
      stage_tile_async(Vs[cur ^ 1], Vbh + (kt + 64), S_);
      S_WAIT_ASYNCCNT(8);  // tile t's 8 ops retired (in-order completion)
    } else {
      S_WAIT_ASYNCCNT(0);
    }
    __syncthreads();  // publish LDS tile to all waves

    // ---- S = Q K^T (8 WMMAs) from LDS ----
    f32x8 sA[4] = {};
    #pragma unroll
    for (int j = 0; j < 4; ++j) {
      bf16x16 kb0 = frag_ld(&Ks[cur][j * 16][0], 72, 0);
      bf16x16 kb1 = frag_ld(&Ks[cur][j * 16][0], 72, 32);
      sA[j] = WMMA(qa0, kb0, sA[j]);
      sA[j] = WMMA(qa1, kb1, sA[j]);
    }
    // ---- scale + ALiBi + mask ----
    float sqn[4]; int ng[4];
    #pragma unroll
    for (int j = 0; j < 4; ++j) {
      ng[j] = kt + j * 16 + nlan;
      sqn[j] = sqrtf((float)ng[j]);
    }
    #pragma unroll
    for (int j = 0; j < 4; ++j) {
      #pragma unroll
      for (int r = 0; r < 8; ++r) {
        float v = sA[j][r] * 0.125f + slope * fabsf(sqn[j] - sqm[r]);
        if (grp == 0 && ng[j] > mg[r]) v = -1e30f;
        if (grp == 1 && ng[j] < mg[r]) v = -1e30f;
        sA[j][r] = v;
      }
    }
    // ---- online softmax (row reductions across the 16-lane half) ----
    #pragma unroll
    for (int r = 0; r < 8; ++r) {
      float x = fmaxf(fmaxf(sA[0][r], sA[1][r]), fmaxf(sA[2][r], sA[3][r]));
      #pragma unroll
      for (int off = 1; off < 16; off <<= 1)
        x = fmaxf(x, __shfl_xor(x, off, 32));
      float mn   = fmaxf(m_r[r], x);
      float corr = __expf(m_r[r] - mn);
      m_r[r] = mn;
      l_r[r] *= corr;
      #pragma unroll
      for (int j = 0; j < 4; ++j) o[j][r] *= corr;
      float rs = 0.0f;
      #pragma unroll
      for (int j = 0; j < 4; ++j) {
        float p = __expf(sA[j][r] - mn);
        sA[j][r] = p;
        rs += p;
      }
      #pragma unroll
      for (int off = 1; off < 16; off <<= 1)
        rs += __shfl_xor(rs, off, 32);
      l_r[r] += rs;
    }
    // ---- P (C layout) -> LDS -> A-fragment layout ----
    #pragma unroll
    for (int j = 0; j < 4; ++j) {
      #pragma unroll
      for (int r = 0; r < 8; ++r)
        Ps[wave][r + halfoff][j * 16 + nlan] = f2bf(sA[j][r]);
    }
    bf16x16 pa0 = frag_ld(&Ps[wave][0][0], 72, 0);
    bf16x16 pa1 = frag_ld(&Ps[wave][0][0], 72, 32);
    // ---- O += P V (8 WMMAs); Vs rows are d, cols are local keys ----
    #pragma unroll
    for (int j = 0; j < 4; ++j) {
      bf16x16 vb0 = frag_ld(&Vs[cur][j * 16][0], 72, 0);
      bf16x16 vb1 = frag_ld(&Vs[cur][j * 16][0], 72, 32);
      o[j] = WMMA(pa0, vb0, o[j]);
      o[j] = WMMA(pa1, vb1, o[j]);
    }
  }

  // ---- epilogue: O / l, write f32 [B,S,H*D] ----
  #pragma unroll
  for (int r = 0; r < 8; ++r) {
    float inv = 1.0f / l_r[r];
    int sg = q0 + r + halfoff;
    float* op = out + ((size_t)b * S_ + sg) * (H_ * D_) + h * D_;
    #pragma unroll
    for (int j = 0; j < 4; ++j)
      op[j * 16 + nlan] = o[j][r] * inv;
  }
}

extern "C" void kernel_launch(void* const* d_in, const int* in_sizes, int n_in,
                              void* d_out, int out_size, void* d_ws, size_t ws_size,
                              hipStream_t stream) {
  const float* X  = (const float*)d_in[0];
  const float* Wq = (const float*)d_in[1];
  const float* bq = (const float*)d_in[2];
  const float* Wk = (const float*)d_in[3];
  const float* bk = (const float*)d_in[4];
  const float* Wv = (const float*)d_in[5];
  const float* bv = (const float*)d_in[6];
  float* out = (float*)d_out;

  const size_t NE = (size_t)B_ * H_ * S_ * D_;  // 3,145,728 elems per tensor
  bf16_t* qws  = (bf16_t*)d_ws;
  bf16_t* kws  = qws + NE;
  bf16_t* vtws = kws + NE;

  dim3 g1((B_ * S_) / 128, E_ / 64, 3);
  qkv_gemm_kernel<<<g1, 256, 0, stream>>>(X, Wq, bq, Wk, bk, Wv, bv,
                                          qws, kws, vtws);

  dim3 g2(S_ / 64, H_, B_);
  alibi_attn_kernel<<<g2, 128, 0, stream>>>(qws, kws, vtws, out);
}
